// GAT_23416161697969
// MI455X (gfx1250) — compile-verified, matrix operand order checked by reading, not run on hardware
//
#include <hip/hip_runtime.h>
#include <hip/hip_bf16.h>

// ---------------------------------------------------------------------------
// 2-layer GAT for MI455X (gfx1250, wave32).
// GEMMs: bf16 WMMA 16x16x32, f32 accumulate. Edge softmax/agg: f32 atomics.
// ---------------------------------------------------------------------------

typedef __attribute__((ext_vector_type(16))) __bf16 v16bf;
typedef __attribute__((ext_vector_type(8)))  __bf16 v8bf;   // 16 bytes
typedef __attribute__((ext_vector_type(8)))  float  v8f;

#define NEG_SLOPE 0.2f

__device__ __forceinline__ __bf16 f32_to_bf16(float f) {
    unsigned u = __float_as_uint(f);
    u += 0x7fffu + ((u >> 16) & 1u);          // round-to-nearest-even
    unsigned short h = (unsigned short)(u >> 16);
    return __builtin_bit_cast(__bf16, h);
}

// order-preserving float<->int encoding for atomicMax-based segment max
__device__ __forceinline__ int f2ord(float x) {
    int b = __float_as_int(x);
    return b < 0 ? (b ^ 0x7fffffff) : b;
}
__device__ __forceinline__ float ord2f(int b) {
    return __int_as_float(b < 0 ? (b ^ 0x7fffffff) : b);
}

// ------------------------------ utility kernels ----------------------------

__global__ void k_zero_f32(float* p, size_t n) {
    size_t i = (size_t)blockIdx.x * blockDim.x + threadIdx.x;
    size_t stride = (size_t)gridDim.x * blockDim.x;
    for (; i < n; i += stride) p[i] = 0.0f;
}

__global__ void k_fill_i32(int* p, int v, size_t n) {
    size_t i = (size_t)blockIdx.x * blockDim.x + threadIdx.x;
    size_t stride = (size_t)gridDim.x * blockDim.x;
    for (; i < n; i += stride) p[i] = v;
}

__global__ void k_f32_to_bf16(const float* __restrict__ x, __bf16* __restrict__ y, size_t n) {
    size_t i = (size_t)blockIdx.x * blockDim.x + threadIdx.x;
    size_t stride = (size_t)gridDim.x * blockDim.x;
    for (; i < n; i += stride) y[i] = f32_to_bf16(x[i]);
}

// W [K][N] row-major f32 -> Wt [N][K] bf16 (so B-fragment columns are contiguous)
__global__ void k_transpose_f32_to_bf16(const float* __restrict__ W, __bf16* __restrict__ Wt,
                                        int K, int N) {
    size_t n = (size_t)K * N;
    size_t i = (size_t)blockIdx.x * blockDim.x + threadIdx.x;
    size_t stride = (size_t)gridDim.x * blockDim.x;
    for (; i < n; i += stride) {
        int k = (int)(i / N);
        int c = (int)(i % N);
        Wt[(size_t)c * K + k] = f32_to_bf16(W[i]);
    }
}

// ------------------------------ WMMA GEMM ----------------------------------
// C[M][N] = A[M][K] (bf16, row-major) x Bt[N][K] (bf16, row-major per column).
// One wave computes one 16x16 tile; K stepped by 32 per v_wmma_f32_16x16x32_bf16.
// A-fragment per lane (ISA 7.12.2, 16-bit A 16x32):
//   lane<16 : row=lane,    K = {k0..k0+7} and {k0+16..k0+23}
//   lane>=16: row=lane-16, K = {k0+8..k0+15} and {k0+24..k0+31}
// B-fragment mirrors this with column index in place of row.
// C/D: VGPR j -> row = j + (lane>=16 ? 8 : 0), col = lane&15.

__global__ void k_gemm_wmma_bf16(const __bf16* __restrict__ A, const __bf16* __restrict__ Bt,
                                 float* __restrict__ C, int M, int N, int K, int totTiles) {
    const int lane = threadIdx.x & 31;
    const int wave = threadIdx.x >> 5;
    const int tile = blockIdx.x * (blockDim.x >> 5) + wave;
    if (tile >= totTiles) return;
    const int tilesN = N >> 4;
    const int bm = tile / tilesN;
    const int bn = tile % tilesN;

    const int r  = lane & 15;
    const int hi = lane >> 4;
    const size_t rowA = (size_t)(bm * 16 + r) * K;
    const size_t colB = (size_t)(bn * 16 + r) * K;

    v8f acc = {};
    for (int k0 = 0; k0 < K; k0 += 32) {
        const int ka = k0 + (hi ? 8 : 0);
        const int kb = k0 + (hi ? 24 : 16);
        v8bf alo = *(const v8bf*)(A + rowA + ka);
        v8bf ahi = *(const v8bf*)(A + rowA + kb);
        v8bf blo = *(const v8bf*)(Bt + colB + ka);
        v8bf bhi = *(const v8bf*)(Bt + colB + kb);
        v16bf av, bv;
#pragma unroll
        for (int i = 0; i < 8; ++i) {
            av[i] = alo[i]; av[i + 8] = ahi[i];
            bv[i] = blo[i]; bv[i + 8] = bhi[i];
        }
        acc = __builtin_amdgcn_wmma_f32_16x16x32_bf16(
            /*neg_a=*/false, av, /*neg_b=*/false, bv,
            /*c_mod=*/(short)0, acc, /*reuse_a=*/false, /*reuse_b=*/false);
    }
#pragma unroll
    for (int j = 0; j < 8; ++j) {
        int row = bm * 16 + j + hi * 8;
        C[(size_t)row * N + bn * 16 + r] = acc[j];
    }
}

// -------------------------- attention scalar dots --------------------------
// feat laid out [N*H, D]; one wave per (node,head): el = feat.al , er = feat.ar
__global__ void k_attn_dot(const float* __restrict__ feat,
                           const float* __restrict__ al, const float* __restrict__ ar,
                           float* __restrict__ el, float* __restrict__ er,
                           int NH, int H, int D) {
    const int lane = threadIdx.x & 31;
    const int w = blockIdx.x * (blockDim.x >> 5) + (threadIdx.x >> 5);
    if (w >= NH) return;
    const int h = w % H;
    const float* f = feat + (size_t)w * D;
    float sl = 0.f, sr = 0.f;
    for (int d = lane; d < D; d += 32) {
        float v = f[d];
        sl += v * al[h * D + d];
        sr += v * ar[h * D + d];
    }
#pragma unroll
    for (int off = 16; off; off >>= 1) {
        sl += __shfl_xor(sl, off, 32);
        sr += __shfl_xor(sr, off, 32);
    }
    if (lane == 0) { el[w] = sl; er[w] = sr; }
}

// ----------------------------- edge softmax --------------------------------

// pass 1: e = leaky_relu(el[src]+er[dst]); stash logits; segment max via int atomicMax
__global__ void k_edge_logits(const int* __restrict__ src, const int* __restrict__ dst,
                              const float* __restrict__ el, const float* __restrict__ er,
                              float* __restrict__ ev, int* __restrict__ menc,
                              int E, int H) {
    size_t n = (size_t)E * H;
    size_t i = (size_t)blockIdx.x * blockDim.x + threadIdx.x;
    size_t stride = (size_t)gridDim.x * blockDim.x;
    for (; i < n; i += stride) {
        int e = (int)(i / H), h = (int)(i % H);
        float v = el[src[e] * H + h] + er[dst[e] * H + h];
        v = v > 0.f ? v : v * NEG_SLOPE;
        ev[i] = v;
        atomicMax(&menc[dst[e] * H + h], f2ord(v));
    }
}

// pass 2: ex = exp(e - m[dst]); segment sum
__global__ void k_edge_exp(const int* __restrict__ dst,
                           float* __restrict__ ev, const int* __restrict__ menc,
                           float* __restrict__ s, int E, int H) {
    size_t n = (size_t)E * H;
    size_t i = (size_t)blockIdx.x * blockDim.x + threadIdx.x;
    size_t stride = (size_t)gridDim.x * blockDim.x;
    for (; i < n; i += stride) {
        int e = (int)(i / H), h = (int)(i % H);
        float ex = __expf(ev[i] - ord2f(menc[dst[e] * H + h]));
        ev[i] = ex;
        atomicAdd(&s[dst[e] * H + h], ex);
    }
}

// pass 3: rst[dst,h,:] += feat[src,h,:] * (ex / s[dst,h]); one wave per (edge,head)
__global__ void k_edge_agg(const int* __restrict__ src, const int* __restrict__ dst,
                           const float* __restrict__ ev, const float* __restrict__ s,
                           const float* __restrict__ feat, float* __restrict__ rst,
                           int E, int H, int D) {
    const int lane = threadIdx.x & 31;
    const int w = blockIdx.x * (blockDim.x >> 5) + (threadIdx.x >> 5);
    if (w >= E * H) return;
    const int e = w / H, h = w % H;
    const int sN = src[e], dN = dst[e];
    const float alpha = ev[w] / s[dN * H + h];
    const float* f = feat + ((size_t)sN * H + h) * D;
    float* r = rst + ((size_t)dN * H + h) * D;
    for (int d = lane; d < D; d += 32)
        atomicAdd(&r[d], f[d] * alpha);
}

// -------------------------- activations / bias -----------------------------

// h1 = elu(rst + b) -> bf16 for layer-2 GEMM
__global__ void k_elu_bias_bf16(const float* __restrict__ rst, const float* __restrict__ b,
                                __bf16* __restrict__ out, size_t n, int C) {
    size_t i = (size_t)blockIdx.x * blockDim.x + threadIdx.x;
    size_t stride = (size_t)gridDim.x * blockDim.x;
    for (; i < n; i += stride) {
        float v = rst[i] + b[i % C];
        v = v > 0.f ? v : (__expf(v) - 1.f);
        out[i] = f32_to_bf16(v);
    }
}

__global__ void k_bias_add(float* __restrict__ out, const float* __restrict__ b,
                           size_t n, int C) {
    size_t i = (size_t)blockIdx.x * blockDim.x + threadIdx.x;
    size_t stride = (size_t)gridDim.x * blockDim.x;
    for (; i < n; i += stride) out[i] += b[i % C];
}

// ------------------------------- launcher ----------------------------------

static inline int gblocks(size_t n, int t = 256) {
    size_t b = (n + t - 1) / t;
    if (b > (size_t)1048576) b = 1048576;
    return (int)b;
}

extern "C" void kernel_launch(void* const* d_in, const int* in_sizes, int n_in,
                              void* d_out, int out_size, void* d_ws, size_t ws_size,
                              hipStream_t stream) {
    const float* x   = (const float*)d_in[0];
    const int*   src = (const int*)d_in[1];
    const int*   dst = (const int*)d_in[2];
    const float* W1  = (const float*)d_in[3];
    const float* al1 = (const float*)d_in[4];
    const float* ar1 = (const float*)d_in[5];
    const float* b1  = (const float*)d_in[6];
    const float* W2  = (const float*)d_in[7];
    const float* al2 = (const float*)d_in[8];
    const float* ar2 = (const float*)d_in[9];
    const float* b2  = (const float*)d_in[10];
    float* out = (float*)d_out;

    const int IN_DIM = 256, HID = 128, OUT = 64;
    const int H0 = 4, H1 = 1;
    const int F1 = H0 * HID;          // 512
    const int N = in_sizes[0] / IN_DIM;
    const int E = in_sizes[1];

    // ---- workspace carve-out (256B aligned) ----
    char* ws = (char*)d_ws;
    size_t cur = 0;
    auto alloc = [&](size_t bytes) {
        void* p = ws + cur;
        cur += (bytes + 255) & ~(size_t)255;
        return p;
    };
    __bf16* xb   = (__bf16*)alloc((size_t)N * IN_DIM * 2);
    __bf16* W1t  = (__bf16*)alloc((size_t)F1 * IN_DIM * 2);
    __bf16* W2t  = (__bf16*)alloc((size_t)OUT * F1 * 2);
    float*  feat1= (float*) alloc((size_t)N * F1 * 4);
    float*  el1  = (float*) alloc((size_t)N * H0 * 4);
    float*  er1  = (float*) alloc((size_t)N * H0 * 4);
    int*    m1   = (int*)   alloc((size_t)N * H0 * 4);
    float*  s1   = (float*) alloc((size_t)N * H0 * 4);
    float*  ev1  = (float*) alloc((size_t)E * H0 * 4);
    float*  rst1 = (float*) alloc((size_t)N * F1 * 4);
    __bf16* h1b  = (__bf16*)alloc((size_t)N * F1 * 2);
    float*  feat2= (float*) alloc((size_t)N * OUT * 4);
    float*  el2  = (float*) alloc((size_t)N * H1 * 4);
    float*  er2  = (float*) alloc((size_t)N * H1 * 4);
    int*    m2   = (int*)   alloc((size_t)N * H1 * 4);
    float*  s2   = (float*) alloc((size_t)N * H1 * 4);
    float*  ev2  = (float*) alloc((size_t)E * H1 * 4);
    (void)ws_size; (void)n_in;

    // ---- init (must re-run every call; harness does not re-poison) ----
    k_zero_f32<<<gblocks((size_t)N * F1), 256, 0, stream>>>(rst1, (size_t)N * F1);
    k_zero_f32<<<gblocks((size_t)N * H0), 256, 0, stream>>>(s1, (size_t)N * H0);
    k_zero_f32<<<gblocks((size_t)N * H1), 256, 0, stream>>>(s2, (size_t)N * H1);
    k_zero_f32<<<gblocks((size_t)out_size), 256, 0, stream>>>(out, (size_t)out_size);
    k_fill_i32<<<gblocks((size_t)N * H0), 256, 0, stream>>>(m1, (int)0x80000000, (size_t)N * H0);
    k_fill_i32<<<gblocks((size_t)N * H1), 256, 0, stream>>>(m2, (int)0x80000000, (size_t)N * H1);

    // ---- precision conversion ----
    k_f32_to_bf16<<<gblocks((size_t)N * IN_DIM), 256, 0, stream>>>(x, xb, (size_t)N * IN_DIM);
    k_transpose_f32_to_bf16<<<gblocks((size_t)IN_DIM * F1), 256, 0, stream>>>(W1, W1t, IN_DIM, F1);
    k_transpose_f32_to_bf16<<<gblocks((size_t)F1 * OUT), 256, 0, stream>>>(W2, W2t, F1, OUT);

    // ---- layer 1: feat1 = x @ W1 (bf16 WMMA, f32 acc) ----
    {
        int tiles = (N / 16) * (F1 / 16);
        k_gemm_wmma_bf16<<<(tiles + 7) / 8, 256, 0, stream>>>(xb, W1t, feat1, N, F1, IN_DIM, tiles);
    }
    k_attn_dot<<<(N * H0 + 7) / 8, 256, 0, stream>>>(feat1, al1, ar1, el1, er1, N * H0, H0, HID);
    k_edge_logits<<<gblocks((size_t)E * H0), 256, 0, stream>>>(src, dst, el1, er1, ev1, m1, E, H0);
    k_edge_exp   <<<gblocks((size_t)E * H0), 256, 0, stream>>>(dst, ev1, m1, s1, E, H0);
    k_edge_agg   <<<(E * H0 + 7) / 8, 256, 0, stream>>>(src, dst, ev1, s1, feat1, rst1, E, H0, HID);

    // ---- ELU(rst1 + b1) -> bf16 ----
    k_elu_bias_bf16<<<gblocks((size_t)N * F1), 256, 0, stream>>>(rst1, b1, h1b, (size_t)N * F1, F1);

    // ---- layer 2: feat2 = h1 @ W2 ----
    {
        int tiles = (N / 16) * (OUT / 16);
        k_gemm_wmma_bf16<<<(tiles + 7) / 8, 256, 0, stream>>>(h1b, W2t, feat2, N, OUT, F1, tiles);
    }
    k_attn_dot<<<(N * H1 + 7) / 8, 256, 0, stream>>>(feat2, al2, ar2, el2, er2, N * H1, H1, OUT);
    k_edge_logits<<<gblocks((size_t)E * H1), 256, 0, stream>>>(src, dst, el2, er2, ev2, m2, E, H1);
    k_edge_exp   <<<gblocks((size_t)E * H1), 256, 0, stream>>>(dst, ev2, m2, s2, E, H1);
    k_edge_agg   <<<(E * H1 + 7) / 8, 256, 0, stream>>>(src, dst, ev2, s2, feat2, out, E, H1, OUT);

    // ---- out += b2 (mean over H1=1 head is identity) ----
    k_bias_add<<<gblocks((size_t)out_size), 256, 0, stream>>>(out, b2, (size_t)out_size, OUT);
}
